// EncoderLayer_73297911873661
// MI455X (gfx1250) — compile-verified, matrix operand order checked by reading, not run on hardware
//
#include <hip/hip_runtime.h>

// Problem constants (fixed by the reference).
#define L_  32768
#define C_  512
#define H_  8
#define D_  64
#define E_  1048576
#define HID_ 1024

typedef __attribute__((ext_vector_type(16))) __bf16 v16bf;
typedef __attribute__((ext_vector_type(8)))  float  v8f;
typedef __attribute__((ext_vector_type(4)))  unsigned int v4u;
typedef __attribute__((ext_vector_type(8)))  int    v8i;
typedef __attribute__((ext_vector_type(4)))  int    v4i;

union Frag16 { v16bf v; uint4 q[2]; unsigned short u[16]; };

static __device__ __forceinline__ unsigned short f2bf(float f) {
    unsigned int u = __float_as_uint(f);
    u += 0x7fffu + ((u >> 16) & 1u);   // round-to-nearest-even
    return (unsigned short)(u >> 16);
}

// ---------------------------------------------------------------- converts
__global__ void f32_to_bf16_kernel(const float* __restrict__ in,
                                   unsigned short* __restrict__ out, int n) {
    for (int i = blockIdx.x * blockDim.x + threadIdx.x; i < n;
         i += gridDim.x * blockDim.x)
        out[i] = f2bf(in[i]);
}

// W[K][N] (f32) -> Wt[N][K] (bf16): makes GEMM B-tiles row-major in N so the
// TDM can stage them exactly like A-tiles and fragments load like A fragments.
__global__ void transpose_f32_to_bf16_kernel(const float* __restrict__ in,
                                             unsigned short* __restrict__ out,
                                             int K, int N) {
    const int total = K * N;
    for (int i = blockIdx.x * blockDim.x + threadIdx.x; i < total;
         i += gridDim.x * blockDim.x) {
        const int k = i / N, n = i - k * N;        // coalesced read
        out[(size_t)n * K + k] = f2bf(in[i]);
    }
}

// ---------------------------------------------------------------- layernorm
__global__ __launch_bounds__(256)
void layernorm_bf16_kernel(const float* __restrict__ x,
                           const float* __restrict__ g,
                           const float* __restrict__ b,
                           unsigned short* __restrict__ z) {
    __shared__ float s1[256], s2[256];
    __shared__ float mu_s, rs_s;
    const int r = blockIdx.x, t = threadIdx.x;
    const float* xr = x + (size_t)r * C_;
    float a0 = xr[t], a1 = xr[t + 256];
    s1[t] = a0 + a1;
    s2[t] = a0 * a0 + a1 * a1;
    __syncthreads();
    for (int st = 128; st > 0; st >>= 1) {
        if (t < st) { s1[t] += s1[t + st]; s2[t] += s2[t + st]; }
        __syncthreads();
    }
    if (t == 0) {
        float mu = s1[0] * (1.0f / C_);
        float var = s2[0] * (1.0f / C_) - mu * mu;
        mu_s = mu;
        rs_s = rsqrtf(var + 1e-5f);
    }
    __syncthreads();
    float mu = mu_s, rs = rs_s;
    unsigned short* zr = z + (size_t)r * C_;
    zr[t]       = f2bf((a0 - mu) * rs * g[t] + b[t]);
    zr[t + 256] = f2bf((a1 - mu) * rs * g[t + 256] + b[t + 256]);
}

// ---------------------------------------------------------------- TDM helper
// 2D tile load: tile_rows x BK bf16 elements from a row-major tensor with row
// length `dim0` elements, into LDS at `lds_off`, with LDS padding of 32B after
// every 64B row (pad_interval=3 -> 16 DWORDs, pad_amount=7 -> 8 DWORDs),
// giving a 96B LDS row stride (= LDA/LDB of 48 ushorts).
static __device__ __forceinline__ void tdm_load_tile(
    unsigned lds_off, const unsigned short* gptr,
    unsigned dim0 /*row len*/, unsigned dim1 /*row count*/,
    unsigned tile_rows) {
    const unsigned long long ga = (unsigned long long)gptr;
    v4u g0;
    g0[0] = 1u;                                    // count=1, user descriptor
    g0[1] = lds_off;                               // LDS byte address
    g0[2] = (unsigned)(ga & 0xffffffffu);          // global_addr[31:0]
    g0[3] = (unsigned)((ga >> 32) & 0x01ffffffu)   // global_addr[56:32]
            | (2u << 30);                          // type = 2 (image)
    v8i g1;
    g1[0] = (int)((1u << 16)                       // data_size = 2 bytes
                | (1u << 20)                       // pad_enable
                | (3u << 22)                       // pad_interval: 16 DWORDs
                | (7u << 25));                     // pad_amount:   8 DWORDs
    g1[1] = (int)((dim0 & 0xffffu) << 16);         // tensor_dim0 lo16
    g1[2] = (int)(((dim0 >> 16) & 0xffffu)         // tensor_dim0 hi16
                | ((dim1 & 0xffffu) << 16));       // tensor_dim1 lo16
    g1[3] = (int)(((dim1 >> 16) & 0xffffu)         // tensor_dim1 hi16
                | (32u << 16));                    // tile_dim0 = BK = 32
    g1[4] = (int)(tile_rows & 0xffffu);            // tile_dim1; tile_dim2 = 0
    g1[5] = (int)dim0;                             // tensor_dim0_stride lo32
    g1[6] = 0;                                     // stride hi16, dim1_stride lo
    g1[7] = 0;
    v4i gz  = {0, 0, 0, 0};                        // groups 2/3: 2D tensor
    v8i gz8 = {0, 0, 0, 0, 0, 0, 0, 0};            // extra group (6-arg form)
    __builtin_amdgcn_tensor_load_to_lds(g0, g1, gz, gz, gz8, 0);
}

// ---------------------------------------------------------------- WMMA GEMM
// Y[M,N] = act( A[M,K](bf16) @ Bt[N,K](bf16)^T + bias[N] ) (+ res[M,N])
// 128x64 tile per 256-thread block (8 waves), BK=32, TDM double-buffered LDS.
#define BM 128
#define BN 64
#define BK 32
#define LDA 48   // ushort units; 96B row stride (TDM pad) keeps b128 alignment
#define LDB 48

template<int ACT, int RES, int OBF>
__global__ __launch_bounds__(256)
void gemm_bf16_kernel(const unsigned short* __restrict__ A,
                      const unsigned short* __restrict__ Bt,
                      const float* __restrict__ bias,
                      const float* __restrict__ res,
                      float* __restrict__ outF,
                      unsigned short* __restrict__ outB,
                      int M, int N, int K) {
    __shared__ unsigned short As[2][BM * LDA];
    __shared__ unsigned short Bs[2][BN * LDB];

    const int t    = threadIdx.x;
    const int wave = t >> 5;
    const int lane = t & 31;
    const int row0 = blockIdx.x * BM;
    const int col0 = blockIdx.y * BN;

    v8f acc[4];
#pragma unroll
    for (int i = 0; i < 4; ++i)
#pragma unroll
        for (int j = 0; j < 8; ++j) acc[i][j] = 0.0f;

    const int m  = lane & 15;
    const int kb = (lane >> 4) << 3;   // 0 or 8 (K-half select, ISA layout)
    const int nsteps = K / BK;

    const unsigned asOff0 = (unsigned)(size_t)&As[0][0];
    const unsigned asOff1 = (unsigned)(size_t)&As[1][0];
    const unsigned bsOff0 = (unsigned)(size_t)&Bs[0][0];
    const unsigned bsOff1 = (unsigned)(size_t)&Bs[1][0];

    if (wave == 0) {
        tdm_load_tile(asOff0, A  + (size_t)row0 * K, K, M, BM);
        tdm_load_tile(bsOff0, Bt + (size_t)col0 * K, K, N, BN);
    }

    for (int s = 0; s < nsteps; ++s) {
        const int cur = s & 1;
        if (wave == 0) {
            if (s + 1 < nsteps) {
                const int k1 = (s + 1) * BK;
                tdm_load_tile(cur ? asOff0 : asOff1,
                              A + (size_t)row0 * K + k1, K, M, BM);
                tdm_load_tile(cur ? bsOff0 : bsOff1,
                              Bt + (size_t)col0 * K + k1, K, N, BN);
                __builtin_amdgcn_s_wait_tensorcnt(2);  // tile s fully landed
            } else {
                __builtin_amdgcn_s_wait_tensorcnt(0);
            }
        }
        __syncthreads();   // publish LDS tile `cur` to all waves

        // A fragment: lane m = row, K pattern {kb..kb+7, 16+kb..16+kb+7}
        Frag16 af;
        const unsigned short* arp = &As[cur][(wave * 16 + m) * LDA];
        af.q[0] = *(const uint4*)(arp + kb);
        af.q[1] = *(const uint4*)(arp + 16 + kb);
#pragma unroll
        for (int nt = 0; nt < 4; ++nt) {
            Frag16 bfrag;
            const unsigned short* brp = &Bs[cur][(nt * 16 + m) * LDB];
            bfrag.q[0] = *(const uint4*)(brp + kb);
            bfrag.q[1] = *(const uint4*)(brp + 16 + kb);
            acc[nt] = __builtin_amdgcn_wmma_f32_16x16x32_bf16(
                false, af.v, false, bfrag.v, (short)0, acc[nt], false, false);
        }
        __syncthreads();   // all readers done before buffer is re-issued
    }

    // Epilogue: D layout — VGPR r holds M = r + 8*(lane/16), N = lane%16
    const int n    = lane & 15;
    const int half = lane >> 4;
    const int gm0  = row0 + wave * 16 + half * 8;
#pragma unroll
    for (int nt = 0; nt < 4; ++nt) {
        const int gn = col0 + nt * 16 + n;
        const float bb = bias[gn];
#pragma unroll
        for (int r = 0; r < 8; ++r) {
            const int gm = gm0 + r;
            float v = acc[nt][r] + bb;
            if (ACT == 1) v = v / (1.0f + __expf(-v));   // SiLU
            if (RES) v += res[(size_t)gm * N + gn];
            if (OBF) outB[(size_t)gm * N + gn] = f2bf(v);
            else     outF[(size_t)gm * N + gn] = v;
        }
    }
}

// ---------------------------------------------------------------- edge scores
// One wave per edge: 8 heads x 4 lanes, each lane dots 16 dims.
__global__ __launch_bounds__(256)
void edge_scores_kernel(const float* __restrict__ Q,
                        const float* __restrict__ Km,
                        const int* __restrict__ row,
                        const int* __restrict__ col,
                        const float* __restrict__ bias,
                        float* __restrict__ scores, int E) {
    const int wave = threadIdx.x >> 5, lane = threadIdx.x & 31;
    const long e = (long)blockIdx.x * 8 + wave;
    if (e >= E) return;
    const int r = row[e], c = col[e];
    const int h = lane >> 2, i = lane & 3;
    const float4* qp = (const float4*)(Q  + (size_t)r * C_ + h * D_ + i * 16);
    const float4* kp = (const float4*)(Km + (size_t)c * C_ + h * D_ + i * 16);
    float s = 0.0f;
#pragma unroll
    for (int j = 0; j < 4; ++j) {
        float4 a = qp[j], b = kp[j];
        s += a.x * b.x + a.y * b.y + a.z * b.z + a.w * b.w;
    }
    s += __shfl_xor(s, 1);
    s += __shfl_xor(s, 2);
    if (i == 0) scores[e * H_ + h] = s * 0.125f + bias[e * H_ + h];
}

// ---------------------------------------------------------------- per-row attn
static __device__ __forceinline__ int lower_bound_i32(const int* a, int n, int key) {
    int lo = 0, hi = n;
    while (lo < hi) {
        int mid = (lo + hi) >> 1;
        if (a[mid] < key) lo = mid + 1; else hi = mid;
    }
    return lo;
}

__global__ __launch_bounds__(256)
void attn_row_kernel(const float* __restrict__ V,
                     const float* __restrict__ scores,
                     const int* __restrict__ row,
                     const int* __restrict__ col,
                     unsigned short* __restrict__ att, int E) {
    __shared__ float red[256];
    __shared__ float sm[H_], ss[H_];
    __shared__ float salpha[256];          // 32 edges x 8 heads
    const int r = blockIdx.x, t = threadIdx.x;
    const int start = lower_bound_i32(row, E, r);
    const int end   = lower_bound_i32(row, E, r + 1);
    const int h = t & 7;

    // segment max per head
    float loc = -3.4e38f;
    for (int e = start + (t >> 3); e < end; e += 32)
        loc = fmaxf(loc, scores[(size_t)e * H_ + h]);
    red[t] = loc; __syncthreads();
    for (int st = 128; st >= 8; st >>= 1) {
        if (t < st) red[t] = fmaxf(red[t], red[t + st]);
        __syncthreads();
    }
    if (t < H_) sm[t] = red[t];
    __syncthreads();

    // segment sum of exp per head
    loc = 0.0f;
    const float mh = sm[h];
    for (int e = start + (t >> 3); e < end; e += 32)
        loc += __expf(scores[(size_t)e * H_ + h] - mh);
    red[t] = loc; __syncthreads();
    for (int st = 128; st >= 8; st >>= 1) {
        if (t < st) red[t] += red[t + st];
        __syncthreads();
    }
    if (t < H_) ss[t] = red[t];
    __syncthreads();

    // weighted gather of V: thread t owns channels t and t+256
    float acc0 = 0.0f, acc1 = 0.0f;
    const int h0 = t >> 6;                 // head for channel t (0..3)
    for (int e0 = start; e0 < end; e0 += 32) {
        const int ne = min(32, end - e0);
        if (t < ne * 8) {
            const int e = e0 + (t >> 3), hh = t & 7;
            salpha[t] = __expf(scores[(size_t)e * H_ + hh] - sm[hh]) / ss[hh];
        }
        __syncthreads();
        for (int j = 0; j < ne; ++j) {
            const int c = col[e0 + j];
            const float* vp = V + (size_t)c * C_;
            acc0 += salpha[j * 8 + h0]     * vp[t];
            acc1 += salpha[j * 8 + h0 + 4] * vp[t + 256];
        }
        __syncthreads();
    }
    att[(size_t)r * C_ + t]       = f2bf(acc0);
    att[(size_t)r * C_ + t + 256] = f2bf(acc1);
}

// ---------------------------------------------------------------- launch
extern "C" void kernel_launch(void* const* d_in, const int* in_sizes, int n_in,
                              void* d_out, int out_size, void* d_ws, size_t ws_size,
                              hipStream_t stream) {
    const float* x        = (const float*)d_in[0];
    const int*   row      = (const int*)d_in[1];
    const int*   col      = (const int*)d_in[2];
    const float* att_bias = (const float*)d_in[3];
    const float* Wq = (const float*)d_in[4];  const float* bq = (const float*)d_in[5];
    const float* Wk = (const float*)d_in[6];  const float* bk = (const float*)d_in[7];
    const float* Wv = (const float*)d_in[8];  const float* bv = (const float*)d_in[9];
    const float* Wo = (const float*)d_in[10]; const float* bo = (const float*)d_in[11];
    const float* ln1_g = (const float*)d_in[12]; const float* ln1_b = (const float*)d_in[13];
    const float* ln2_g = (const float*)d_in[14]; const float* ln2_b = (const float*)d_in[15];
    const float* W1 = (const float*)d_in[16]; const float* b1 = (const float*)d_in[17];
    const float* W2 = (const float*)d_in[18]; const float* b2 = (const float*)d_in[19];
    float* out = (float*)d_out;

    // Workspace carve-out
    char* p = (char*)d_ws;
    auto bump = [&](size_t bytes) {
        char* q = p;
        p += (bytes + 255) & ~(size_t)255;
        return q;
    };
    unsigned short* zbf   = (unsigned short*)bump((size_t)L_ * C_ * 2);
    unsigned short* wqb   = (unsigned short*)bump((size_t)C_ * C_ * 2);   // transposed
    unsigned short* wkb   = (unsigned short*)bump((size_t)C_ * C_ * 2);
    unsigned short* wvb   = (unsigned short*)bump((size_t)C_ * C_ * 2);
    unsigned short* wob   = (unsigned short*)bump((size_t)C_ * C_ * 2);
    unsigned short* w1b   = (unsigned short*)bump((size_t)C_ * HID_ * 2);
    unsigned short* w2b   = (unsigned short*)bump((size_t)HID_ * C_ * 2);
    float*          Qf    = (float*)bump((size_t)L_ * C_ * 4);
    float*          Kf    = (float*)bump((size_t)L_ * C_ * 4);
    float*          Vf    = (float*)bump((size_t)L_ * C_ * 4);
    float*          sc    = (float*)bump((size_t)E_ * H_ * 4);
    unsigned short* attb  = (unsigned short*)bump((size_t)L_ * C_ * 2);
    float*          x1    = (float*)bump((size_t)L_ * C_ * 4);
    unsigned short* z2b   = (unsigned short*)bump((size_t)L_ * C_ * 2);
    unsigned short* hmidb = (unsigned short*)bump((size_t)L_ * HID_ * 2);

    // 1) weights -> bf16, transposed to [N][K]
    transpose_f32_to_bf16_kernel<<<512, 256, 0, stream>>>(Wq, wqb, C_, C_);
    transpose_f32_to_bf16_kernel<<<512, 256, 0, stream>>>(Wk, wkb, C_, C_);
    transpose_f32_to_bf16_kernel<<<512, 256, 0, stream>>>(Wv, wvb, C_, C_);
    transpose_f32_to_bf16_kernel<<<512, 256, 0, stream>>>(Wo, wob, C_, C_);
    transpose_f32_to_bf16_kernel<<<1024, 256, 0, stream>>>(W1, w1b, C_, HID_);
    transpose_f32_to_bf16_kernel<<<1024, 256, 0, stream>>>(W2, w2b, HID_, C_);

    // 2) LN1 -> bf16
    layernorm_bf16_kernel<<<L_, 256, 0, stream>>>(x, ln1_g, ln1_b, zbf);

    // 3) Q/K/V projections (TDM-staged WMMA bf16 -> f32)
    dim3 gQKV(L_ / BM, C_ / BN);
    gemm_bf16_kernel<0, 0, 0><<<gQKV, 256, 0, stream>>>(zbf, wqb, bq, nullptr, Qf, nullptr, L_, C_, C_);
    gemm_bf16_kernel<0, 0, 0><<<gQKV, 256, 0, stream>>>(zbf, wkb, bk, nullptr, Kf, nullptr, L_, C_, C_);
    gemm_bf16_kernel<0, 0, 0><<<gQKV, 256, 0, stream>>>(zbf, wvb, bv, nullptr, Vf, nullptr, L_, C_, C_);

    // 4) per-edge scores
    edge_scores_kernel<<<E_ / 8, 256, 0, stream>>>(Qf, Kf, row, col, att_bias, sc, E_);

    // 5) per-row segment softmax + weighted V gather
    attn_row_kernel<<<L_, 256, 0, stream>>>(Vf, sc, row, col, attb, E_);

    // 6) O-projection + residual -> x1
    gemm_bf16_kernel<0, 1, 0><<<gQKV, 256, 0, stream>>>(attb, wob, bo, x, x1, nullptr, L_, C_, C_);

    // 7) LN2 -> bf16
    layernorm_bf16_kernel<<<L_, 256, 0, stream>>>(x1, ln2_g, ln2_b, z2b);

    // 8) MLP up (SiLU, bf16 out)
    dim3 gUP(L_ / BM, HID_ / BN);
    gemm_bf16_kernel<1, 0, 1><<<gUP, 256, 0, stream>>>(z2b, w1b, b1, nullptr, nullptr, hmidb, L_, HID_, C_);

    // 9) MLP down + residual -> d_out (f32)
    dim3 gDN(L_ / BM, C_ / BN);
    gemm_bf16_kernel<0, 1, 0><<<gDN, 256, 0, stream>>>(hmidb, w2b, b2, x1, out, nullptr, L_, C_, HID_);

    (void)n_in; (void)in_sizes; (void)out_size; (void)ws_size;
}
